// MoEAttentionPooling_77970836291866
// MI455X (gfx1250) — matrix operand order repaired and checked by reading, not compiled, not run on hardware
//
#include <hip/hip_runtime.h>

typedef unsigned short u16;
typedef __attribute__((ext_vector_type(16))) __bf16 bf16x16;
typedef __attribute__((ext_vector_type(8)))  float  f32x8;

__device__ __forceinline__ u16 f2b(float f) {
  unsigned u = __builtin_bit_cast(unsigned, f);
  u += 0x7FFFu + ((u >> 16) & 1u);           // round-to-nearest-even
  return (u16)(u >> 16);
}
__device__ __forceinline__ __bf16 asbf(u16 v) {
  return __builtin_bit_cast(__bf16, v);
}

// CDNA5 async copy: 16B global -> LDS, tracked by ASYNCcnt (no VGPR staging).
__device__ __forceinline__ void async_cp16(unsigned lds_off, const u16* gaddr) {
  asm volatile("global_load_async_to_lds_b128 %0, %1, off"
               :: "v"(lds_off), "v"(gaddr) : "memory");
}
__device__ __forceinline__ void wait_async0() {
  asm volatile("s_wait_asynccnt 0x0" ::: "memory");
}

// ---------------------------------------------------------------------------
// Generic bf16 WMMA GEMM:  C[M,N] = A[M,K] * Bp + bias[N]   (f32 accum)
//   A  : row-major bf16 [M x K]
//   Bp : fragment-packed bf16: Bp[((k>>5)*N + n)*32 + (k&31)]
// 128x128 macro tile, BK=32, 256 threads = 8 waves (2x4), wave tile 64x32.
// Double-buffered LDS, async global->LDS prefetch overlapped with WMMA.
// ---------------------------------------------------------------------------
#define BM 128
#define BN 128
#define BK 32
#define SLA 48      // padded LDS stride (elements); 96B, 16B-aligned rows
#define SLB 48

__global__ __launch_bounds__(256) void k_gemm_bf16(
    const u16* __restrict__ A, long long sAb,
    const u16* __restrict__ B, long long sBb,
    const float* __restrict__ bias, long long sBiasb,
    float* __restrict__ C, long long sCb,
    int M, int N, int Kd)
{
  __shared__ __align__(16) u16 shA[2][BM * SLA];
  __shared__ __align__(16) u16 shB[2][BN * SLB];
  const int tid = threadIdx.x;
  const int bz  = blockIdx.z;
  A += (long long)bz * sAb;
  B += (long long)bz * sBb;
  C += (long long)bz * sCb;
  const float* bp = bias ? bias + (long long)bz * sBiasb : nullptr;
  const int m0 = blockIdx.y * BM;
  const int n0 = blockIdx.x * BN;
  const int wid = tid >> 5, lane = tid & 31;
  const int wm = (wid >> 2) * 64;
  const int wn = (wid & 3) * 32;
  const int lm = lane & 15, lg = lane >> 4;

  // per-thread staging chunk coords (identical mapping for A and packed-B)
  const int c0 = tid,        r0 = c0 >> 2, o0 = (c0 & 3) << 3;
  const int c1 = tid + 256,  r1 = c1 >> 2, o1 = (c1 & 3) << 3;

  f32x8 acc[4][2];
  #pragma unroll
  for (int i = 0; i < 4; ++i)
    #pragma unroll
    for (int j = 0; j < 2; ++j)
      #pragma unroll
      for (int r = 0; r < 8; ++r) acc[i][j][r] = 0.0f;

  const int nIter = Kd / BK;

  auto issue = [&](int it, int buf) {
    const int k0 = it * BK;
    const u16* Bt = B + ((size_t)(k0 >> 5) * N + n0) * 32;   // contiguous 8KB tile
    async_cp16((unsigned)(uintptr_t)&shA[buf][r0 * SLA + o0],
               A + (size_t)(m0 + r0) * Kd + k0 + o0);
    async_cp16((unsigned)(uintptr_t)&shA[buf][r1 * SLA + o1],
               A + (size_t)(m0 + r1) * Kd + k0 + o1);
    async_cp16((unsigned)(uintptr_t)&shB[buf][r0 * SLB + o0], Bt + c0 * 8);
    async_cp16((unsigned)(uintptr_t)&shB[buf][r1 * SLB + o1], Bt + c1 * 8);
  };

  issue(0, 0);
  for (int it = 0; it < nIter; ++it) {
    const int cur = it & 1;
    wait_async0();        // this tile's async writes (ours) have landed
    __syncthreads();      // all waves: tile ready; other buffer free for reuse
    if (it + 1 < nIter) issue(it + 1, cur ^ 1);   // overlaps with compute below

    // A fragments: lane (m=lm, half lg) reads k = 16*lg + 0..15 (one contiguous 32B run)
    bf16x16 fa[4], fb[2];
    #pragma unroll
    for (int i = 0; i < 4; ++i) {
      const u16* rp = &shA[cur][(wm + i * 16 + lm) * SLA];
      #pragma unroll
      for (int v = 0; v < 8; ++v) {
        int kb = (lg << 4) + (v << 1);
        fa[i][2 * v]     = asbf(rp[kb]);
        fa[i][2 * v + 1] = asbf(rp[kb + 1]);
      }
    }
    // B fragments: lane (n=lm, half lg) reads 16 contiguous u16 (32B)
    #pragma unroll
    for (int j = 0; j < 2; ++j) {
      const u16* cp = &shB[cur][(wn + j * 16 + lm) * SLB + lg * 16];
      #pragma unroll
      for (int jj = 0; jj < 16; ++jj)
        fb[j][jj] = asbf(cp[jj]);
    }

    #pragma unroll
    for (int i = 0; i < 4; ++i)
      #pragma unroll
      for (int j = 0; j < 2; ++j)
        acc[i][j] = __builtin_amdgcn_wmma_f32_16x16x32_bf16(
            false, fa[i], false, fb[j], (short)0, acc[i][j], false, false);
  }

  // hoisted bias: one value per j-column tile, invariant over i and r
  float bv[2] = {0.f, 0.f};
  if (bp) {
    #pragma unroll
    for (int j = 0; j < 2; ++j) bv[j] = bp[n0 + wn + j * 16 + lm];
  }

  // C/D layout: VGPR r -> M = r + 8*laneGroup, lane%16 -> N
  #pragma unroll
  for (int i = 0; i < 4; ++i)
    #pragma unroll
    for (int j = 0; j < 2; ++j)
      #pragma unroll
      for (int r = 0; r < 8; ++r) {
        int row = m0 + wm + i * 16 + lg * 8 + r;
        int col = n0 + wn + j * 16 + lm;
        C[(size_t)row * N + col] = acc[i][j][r] + bv[j];
      }
}

// ---------------------------------------------------------------------------
// Conversion / packing kernels
// ---------------------------------------------------------------------------
__global__ void k_f32_to_bf16(const float* __restrict__ in, u16* __restrict__ out, size_t n) {
  size_t i = (size_t)blockIdx.x * blockDim.x + threadIdx.x;
  if (i < n) out[i] = f2b(in[i]);
}

// Pack row-major f32 [batch x Kd x N] into WMMA-fragment layout per batch:
//   out[e*Kd*N + ((k>>5)*N + n)*32 + (k&31)] = bf16(in[e,k,n])
__global__ void k_pack_b(const float* __restrict__ in, u16* __restrict__ out,
                         int Kd, int N, size_t total) {
  size_t i = (size_t)blockIdx.x * blockDim.x + threadIdx.x;
  if (i >= total) return;
  size_t kn = (size_t)Kd * N;
  size_t e = i / kn, r = i - e * kn;
  int k = (int)(r / N), n = (int)(r % N);
  out[e * kn + ((size_t)(k >> 5) * N + n) * 32 + (k & 31)] = f2b(in[i]);
}

// Transpose + pack: logical B[k][n] = in[(row_off + n)*ldin + k]
__global__ void k_pack_b_t(const float* __restrict__ in, int row_off, int ldin,
                           u16* __restrict__ out, int Kd, int N) {
  size_t i = (size_t)blockIdx.x * blockDim.x + threadIdx.x;
  if (i >= (size_t)Kd * N) return;
  int k = (int)(i / N), n = (int)(i % N);
  out[((size_t)(k >> 5) * N + n) * 32 + (k & 31)] =
      f2b(in[(size_t)(row_off + n) * ldin + k]);
}

// q[t, dq] = probe[t,:] . in_proj_w[dq,:] + in_proj_b[dq]  (tiny, f32 VALU)
__global__ void k_qproj(const float* __restrict__ probe, const float* __restrict__ w,
                        const float* __restrict__ b, float* __restrict__ q) {
  int t = blockIdx.y;
  int dq = blockIdx.x * 256 + threadIdx.x;
  const float* pr = probe + (size_t)t * 1024;
  const float* wr = w + (size_t)dq * 1024;
  float s = 0.f;
  for (int k = 0; k < 1024; ++k) s += pr[k] * wr[k];
  q[t * 1024 + dq] = s + b[dq];
}

// ---------------------------------------------------------------------------
// Attention: one block per (b,h). q is 8x64, K/V rows pulled from kv buffer
// kv layout: [b*S + s][2048]  (cols 0..1023 = K features, 1024..2047 = V)
// ---------------------------------------------------------------------------
__global__ __launch_bounds__(256) void k_attn(
    const float* __restrict__ q, const float* __restrict__ kv,
    float* __restrict__ ctx, float* __restrict__ attn_full)
{
  __shared__ float sq[512];
  __shared__ float sc[8 * 1024];
  __shared__ float red[256];
  int bh = blockIdx.x, b = bh >> 4, h = bh & 15;
  int tid = threadIdx.x;
  for (int i = tid; i < 512; i += 256) {
    int t = i >> 6, d = i & 63;
    sq[i] = q[t * 1024 + h * 64 + d];
  }
  __syncthreads();
  for (int s = tid; s < 1024; s += 256) {
    const float* kp = kv + ((size_t)(b * 1024 + s)) * 2048 + h * 64;
    float a[8];
    #pragma unroll
    for (int t = 0; t < 8; ++t) a[t] = 0.f;
    for (int d = 0; d < 64; ++d) {
      float kd = kp[d];
      #pragma unroll
      for (int t = 0; t < 8; ++t) a[t] += sq[t * 64 + d] * kd;
    }
    #pragma unroll
    for (int t = 0; t < 8; ++t) sc[t * 1024 + s] = a[t] * 0.125f;  // 1/sqrt(64)
  }
  __syncthreads();
  for (int t = 0; t < 8; ++t) {
    float m = -3.4e38f;
    for (int s = tid; s < 1024; s += 256) m = fmaxf(m, sc[t * 1024 + s]);
    red[tid] = m; __syncthreads();
    for (int st = 128; st > 0; st >>= 1) {
      if (tid < st) red[tid] = fmaxf(red[tid], red[tid + st]);
      __syncthreads();
    }
    m = red[0]; __syncthreads();
    float sum = 0.f;
    for (int s = tid; s < 1024; s += 256) {
      float e = expf(sc[t * 1024 + s] - m);
      sc[t * 1024 + s] = e;
      sum += e;
    }
    red[tid] = sum; __syncthreads();
    for (int st = 128; st > 0; st >>= 1) {
      if (tid < st) red[tid] += red[tid + st];
      __syncthreads();
    }
    float inv = 1.0f / red[0];
    __syncthreads();
    for (int s = tid; s < 1024; s += 256) {
      float a = sc[t * 1024 + s] * inv;
      sc[t * 1024 + s] = a;
      attn_full[((((size_t)b * 16 + h) * 8) + t) * 1024 + s] = a;
    }
    __syncthreads();
  }
  for (int o = tid; o < 512; o += 256) {
    int t = o >> 6, d = o & 63;
    const float* vp = kv + (size_t)b * 1024 * 2048 + 1024 + h * 64 + d;
    float a = 0.f;
    for (int s = 0; s < 1024; ++s) a += sc[t * 1024 + s] * vp[(size_t)s * 2048];
    ctx[((size_t)b * 8 + t) * 1024 + h * 64 + d] = a;
  }
}

// attn_weights[b,t,s] = mean over heads
__global__ void k_attnw(const float* __restrict__ attn_full, float* __restrict__ out) {
  size_t i = (size_t)blockIdx.x * blockDim.x + threadIdx.x;
  if (i >= (size_t)32 * 8 * 1024) return;
  size_t b = i / 8192, r = i % 8192;
  float s = 0.f;
  for (int h = 0; h < 16; ++h) s += attn_full[(b * 16 + h) * 8192 + r];
  out[i] = s * (1.0f / 16.0f);
}

// ---------------------------------------------------------------------------
// LayerNorm + gate logits + softmax + top-2 gating. One block per token n.
// ---------------------------------------------------------------------------
__global__ __launch_bounds__(256) void k_ln_gate(
    const float* __restrict__ attn_out, const float* __restrict__ gamma,
    const float* __restrict__ beta, const float* __restrict__ gate_w,
    u16* __restrict__ y_b, float* __restrict__ gates,
    float* __restrict__ probs, int* __restrict__ topi)
{
  __shared__ float red[256];
  __shared__ float sy[1024];
  __shared__ float rede[8 * 256];
  int n = blockIdx.x, t = n & 7, tid = threadIdx.x;
  const float* xr = attn_out + (size_t)n * 1024;
  float loc[4], s = 0.f;
  #pragma unroll
  for (int i = 0; i < 4; ++i) { loc[i] = xr[tid + 256 * i]; s += loc[i]; }
  red[tid] = s; __syncthreads();
  for (int st = 128; st > 0; st >>= 1) { if (tid < st) red[tid] += red[tid + st]; __syncthreads(); }
  float mean = red[0] * (1.0f / 1024.0f);
  __syncthreads();
  s = 0.f;
  #pragma unroll
  for (int i = 0; i < 4; ++i) { float d = loc[i] - mean; s += d * d; }
  red[tid] = s; __syncthreads();
  for (int st = 128; st > 0; st >>= 1) { if (tid < st) red[tid] += red[tid + st]; __syncthreads(); }
  float inv = rsqrtf(red[0] * (1.0f / 1024.0f) + 1e-5f);
  __syncthreads();
  #pragma unroll
  for (int i = 0; i < 4; ++i) {
    int idx = tid + 256 * i;
    float yv = (loc[i] - mean) * inv * gamma[idx] + beta[idx];
    sy[idx] = yv;
    y_b[(size_t)n * 1024 + idx] = f2b(yv);
  }
  __syncthreads();
  float part[8];
  #pragma unroll
  for (int e = 0; e < 8; ++e) part[e] = 0.f;
  for (int i = tid; i < 1024; i += 256) {
    float yv = sy[i];
    const float* gw = gate_w + ((size_t)t * 1024 + i) * 8;
    #pragma unroll
    for (int e = 0; e < 8; ++e) part[e] += yv * gw[e];
  }
  #pragma unroll
  for (int e = 0; e < 8; ++e) rede[e * 256 + tid] = part[e];
  __syncthreads();
  if (tid == 0) {
    float lg[8];
    for (int e = 0; e < 8; ++e) {
      float acc = 0.f;
      for (int j = 0; j < 256; ++j) acc += rede[e * 256 + j];
      lg[e] = acc;
    }
    float m = lg[0];
    for (int e = 1; e < 8; ++e) m = fmaxf(m, lg[e]);
    float ex[8], sum = 0.f;
    for (int e = 0; e < 8; ++e) { ex[e] = expf(lg[e] - m); sum += ex[e]; }
    float invs = 1.0f / sum;
    float p[8];
    for (int e = 0; e < 8; ++e) { p[e] = ex[e] * invs; probs[n * 8 + e] = p[e]; }
    int i1 = 0;
    for (int e = 1; e < 8; ++e) if (p[e] > p[i1]) i1 = e;
    int i2 = (i1 == 0) ? 1 : 0;
    for (int e = 0; e < 8; ++e) if (e != i1 && p[e] > p[i2]) i2 = e;
    float invw = 1.0f / (p[i1] + p[i2]);
    for (int e = 0; e < 8; ++e) gates[n * 8 + e] = 0.f;
    gates[n * 8 + i1] = p[i1] * invw;
    gates[n * 8 + i2] = p[i2] * invw;
    topi[n * 2] = i1; topi[n * 2 + 1] = i2;
  }
}

__global__ void k_moe_loss(const float* __restrict__ probs, const int* __restrict__ topi,
                           float* __restrict__ out) {
  __shared__ float aP[8], aF[8];
  int tid = threadIdx.x;
  if (tid < 8) { aP[tid] = 0.f; aF[tid] = 0.f; }
  __syncthreads();
  #pragma unroll
  for (int e = 0; e < 8; ++e) atomicAdd(&aP[e], probs[tid * 8 + e]);
  atomicAdd(&aF[topi[tid * 2]], 1.0f);
  atomicAdd(&aF[topi[tid * 2 + 1]], 1.0f);
  __syncthreads();
  if (tid == 0) {
    float l = 0.f;
    for (int e = 0; e < 8; ++e) l += (aF[e] / 256.0f) * (aP[e] / 256.0f);
    out[0] = 8.0f * l;
  }
}

__global__ void k_gelu_bf16(const float* __restrict__ in, u16* __restrict__ out, size_t n) {
  size_t i = (size_t)blockIdx.x * blockDim.x + threadIdx.x;
  if (i >= n) return;
  float x = in[i];
  out[i] = f2b(0.5f * x * (1.0f + erff(x * 0.70710678118654752f)));
}

// final = attn_out + sum_e gates[n,e] * eo[e,n,:]
__global__ void k_final(const float* __restrict__ attn_out, const float* __restrict__ gates,
                        const float* __restrict__ eo, float* __restrict__ out) {
  size_t i = (size_t)blockIdx.x * blockDim.x + threadIdx.x;
  if (i >= (size_t)256 * 1024) return;
  int n = (int)(i >> 10);
  int d = (int)(i & 1023);
  float acc = attn_out[i];
  #pragma unroll
  for (int e = 0; e < 8; ++e) {
    float g = gates[n * 8 + e];
    acc += g * eo[((size_t)e * 256 + n) * 1024 + d];
  }
  out[i] = acc;
}

// ---------------------------------------------------------------------------
extern "C" void kernel_launch(void* const* d_in, const int* in_sizes, int n_in,
                              void* d_out, int out_size, void* d_ws, size_t ws_size,
                              hipStream_t stream) {
  const float* x      = (const float*)d_in[0];
  const float* probe  = (const float*)d_in[1];
  const float* in_w   = (const float*)d_in[2];
  const float* in_b   = (const float*)d_in[3];
  const float* out_w  = (const float*)d_in[4];
  const float* out_b  = (const float*)d_in[5];
  const float* gamma  = (const float*)d_in[6];
  const float* beta   = (const float*)d_in[7];
  const float* gate_w = (const float*)d_in[8];
  const float* w1     = (const float*)d_in[9];
  const float* b1     = (const float*)d_in[10];
  const float* w2     = (const float*)d_in[11];
  const float* b2     = (const float*)d_in[12];
  (void)in_sizes; (void)n_in; (void)out_size; (void)ws_size;

  char* ws = (char*)d_ws;
  size_t off = 0;
  auto alloc = [&](size_t bytes) -> void* {
    void* p = ws + off;
    off += (bytes + 255) & ~(size_t)255;
    return p;
  };
  u16*   x_b    = (u16*)  alloc((size_t)32 * 1024 * 1024 * 2);   // x bf16 (A, row-major)
  u16*   wkv_p  = (u16*)  alloc((size_t)1024 * 2048 * 2);        // [Wk;Wv]^T packed
  u16*   wo_p   = (u16*)  alloc((size_t)1024 * 1024 * 2);        // Wout^T packed
  u16*   w1_p   = (u16*)  alloc((size_t)8 * 1024 * 4096 * 2);    // w1 packed
  u16*   w2_p   = (u16*)  alloc((size_t)8 * 4096 * 1024 * 2);    // w2 packed
  float* qv     = (float*)alloc((size_t)8 * 1024 * 4);
  float* kv     = (float*)alloc((size_t)32768 * 2048 * 4);       // K|V per token
  float* attn_f = (float*)alloc((size_t)32 * 16 * 8 * 1024 * 4);
  float* ctx    = (float*)alloc((size_t)256 * 1024 * 4);
  u16*   ctx_b  = (u16*)  alloc((size_t)256 * 1024 * 2);
  float* attn_o = (float*)alloc((size_t)256 * 1024 * 4);
  u16*   y_b    = (u16*)  alloc((size_t)256 * 1024 * 2);
  float* gates  = (float*)alloc((size_t)256 * 8 * 4);
  float* probs  = (float*)alloc((size_t)256 * 8 * 4);
  int*   topi   = (int*)  alloc((size_t)256 * 2 * 4);
  float* h_f    = (float*)alloc((size_t)8 * 256 * 4096 * 4);
  u16*   h_b    = (u16*)  alloc((size_t)8 * 256 * 4096 * 2);
  float* eo     = (float*)alloc((size_t)8 * 256 * 1024 * 4);

  float* out_final = (float*)d_out;
  float* out_loss  = out_final + 262144;
  float* out_attnw = out_final + 262145;

  // one-time conversions: A-side plain bf16, B-side fragment-packed bf16
  k_f32_to_bf16<<<131072, 256, 0, stream>>>(x, x_b, (size_t)33554432);
  k_pack_b_t<<<8192, 256, 0, stream>>>(in_w, 1024, 1024, wkv_p, 1024, 2048);
  k_pack_b_t<<<4096, 256, 0, stream>>>(out_w, 0, 1024, wo_p, 1024, 1024);
  k_pack_b<<<131072, 256, 0, stream>>>(w1, w1_p, 1024, 4096, (size_t)33554432);
  k_pack_b<<<131072, 256, 0, stream>>>(w2, w2_p, 4096, 1024, (size_t)33554432);

  // Q projection (probe is batch-broadcast -> compute once, f32)
  k_qproj<<<dim3(4, 8), 256, 0, stream>>>(probe, in_w, in_b, qv);

  // fused K/V projection: [32768 x 1024] x [1024 x 2048] + bias
  k_gemm_bf16<<<dim3(16, 256, 1), 256, 0, stream>>>(
      x_b, 0, wkv_p, 0, in_b + 1024, 0, kv, 0, 32768, 2048, 1024);

  // attention per (b,h), softmax in LDS
  k_attn<<<512, 256, 0, stream>>>(qv, kv, ctx, attn_f);
  k_attnw<<<1024, 256, 0, stream>>>(attn_f, out_attnw);

  // out_proj
  k_f32_to_bf16<<<1024, 256, 0, stream>>>(ctx, ctx_b, (size_t)262144);
  k_gemm_bf16<<<dim3(8, 2, 1), 256, 0, stream>>>(
      ctx_b, 0, wo_p, 0, out_b, 0, attn_o, 0, 256, 1024, 1024);

  // LN + gating + aux loss
  k_ln_gate<<<256, 256, 0, stream>>>(attn_o, gamma, beta, gate_w, y_b, gates, probs, topi);
  k_moe_loss<<<1, 256, 0, stream>>>(probs, topi, out_loss);

  // MoE (dense over experts, matching reference): GEMM1 -> gelu -> GEMM2
  k_gemm_bf16<<<dim3(32, 2, 8), 256, 0, stream>>>(
      y_b, 0, w1_p, (long long)1024 * 4096, b1, 4096,
      h_f, (long long)256 * 4096, 256, 4096, 1024);
  k_gelu_bf16<<<32768, 256, 0, stream>>>(h_f, h_b, (size_t)8388608);
  k_gemm_bf16<<<dim3(8, 2, 8), 256, 0, stream>>>(
      h_b, (long long)256 * 4096, w2_p, (long long)4096 * 1024, b2, 1024,
      eo, (long long)256 * 1024, 256, 1024, 4096);

  // gated combine + residual
  k_final<<<1024, 256, 0, stream>>>(attn_o, gates, eo, out_final);
}
